// MotionMamba_24816321036299
// MI455X (gfx1250) — matrix operand order repaired and checked by reading, not compiled, not run on hardware
//
#include <hip/hip_runtime.h>
#include <hip/hip_bf16.h>

// ---------------------------------------------------------------------------
// MotionMamba forward for MI455X (gfx1250)
// GEMMs: fp32->bf16 pre-pack, then v_wmma_f32_16x16x32_bf16 with pure b128 loads
// ---------------------------------------------------------------------------

#define F_DIM   81
#define NJ_DIM  17
#define C_DIM   256
#define DIN     512     // D_INNER
#define DSTATE  16
#define T_TOK   (F_DIM * NJ_DIM)   // 1377 tokens in either view
#define T_PAD   1392               // 87 tiles of 16
#define M_TILES 87

typedef __bf16 v16bf __attribute__((ext_vector_type(16)));
typedef __bf16 v8bf  __attribute__((ext_vector_type(8)));
typedef float  v8f   __attribute__((ext_vector_type(8)));

__device__ __forceinline__ float act_f(float x, int act) {
    if (act == 1) return x / (1.f + __expf(-x));                       // silu
    if (act == 2) {                                                    // gelu (tanh)
        float x3 = x * x * x;
        return 0.5f * x * (1.f + tanhf(0.7978845608f * (x + 0.044715f * x3)));
    }
    if (act == 3) return (x > 20.f) ? x : log1pf(__expf(x));           // softplus
    return x;
}

// ---------------------------------------------------------------------------
// fp32 -> bf16 pack with optional zero K-padding:
// out[r*ldo + c] = (c < Kr) ? bf16(in[r*ldi + c]) : 0      for c in [0, ldo)
// ---------------------------------------------------------------------------
__global__ void k_cvt(const float* __restrict__ in, int ldi,
                      __bf16* __restrict__ out, int ldo, int Kr)
{
    const int r = blockIdx.x;
    for (int c = threadIdx.x; c < ldo; c += blockDim.x) {
        float v = (c < Kr) ? in[r * ldi + c] : 0.f;
        out[r * ldo + c] = (__bf16)v;
    }
}

// ---------------------------------------------------------------------------
// WMMA GEMM on pre-packed bf16:  Out[M,N] = act(A @ W^T + bias) + Res
// One wave per 16(M) x NT*16(N) strip. No bounds checks anywhere:
//  - grid covers exactly N (multiple of NT*16) and M_TILES*16 <= T_PAD rows
//  - K (padded) is a multiple of 32
// Fragments are contiguous bf16 runs -> 2 x b128 loads each.
// ---------------------------------------------------------------------------
template <int NT>
__global__ void k_gemm_bf(const __bf16* __restrict__ A, int lda,
                          const __bf16* __restrict__ W, int ldw,
                          const float* __restrict__ bias,
                          const float* __restrict__ Res, int ldr,
                          float* __restrict__ Out, int ldo,
                          int K, int act)
{
    const int lane = threadIdx.x;      // wave32
    const int half = lane >> 4;        // 0: lanes 0-15, 1: lanes 16-31
    const int l16  = lane & 15;
    const int m    = blockIdx.x * 16 + l16;
    const int nb   = blockIdx.y * (NT * 16);

    v8f acc[NT];
#pragma unroll
    for (int nt = 0; nt < NT; ++nt)
#pragma unroll
        for (int r = 0; r < 8; ++r) acc[nt][r] = 0.f;

    const __bf16* __restrict__ Arow = A + (size_t)m * lda;
    const __bf16* Wp[NT];
#pragma unroll
    for (int nt = 0; nt < NT; ++nt)
        Wp[nt] = W + (size_t)(nb + nt * 16 + l16) * ldw;

    for (int k0 = 0; k0 < K; k0 += 32) {
        if (k0 + 64 < K)
            __builtin_prefetch((const void*)(Arow + k0 + 64), 0, 1);
        // A fragment: a[0..7] = A[m][k0+half*8 ..], a[8..15] = A[m][k0+16+half*8 ..]
        v8bf a0 = *(const v8bf*)(Arow + k0 + half * 8);
        v8bf a1 = *(const v8bf*)(Arow + k0 + 16 + half * 8);
        v16bf av = __builtin_shufflevector(a0, a1,
            0, 1, 2, 3, 4, 5, 6, 7, 8, 9, 10, 11, 12, 13, 14, 15);
#pragma unroll
        for (int nt = 0; nt < NT; ++nt) {
            // B fragment: b[0..7] = W[n][k0+half*8 ..], b[8..15] = W[n][k0+16+half*8 ..]
            v8bf b0 = *(const v8bf*)(Wp[nt] + k0 + half * 8);
            v8bf b1 = *(const v8bf*)(Wp[nt] + k0 + 16 + half * 8);
            v16bf bv = __builtin_shufflevector(b0, b1,
                0, 1, 2, 3, 4, 5, 6, 7, 8, 9, 10, 11, 12, 13, 14, 15);
            acc[nt] = __builtin_amdgcn_wmma_f32_16x16x32_bf16(
                false, av, false, bv, (short)0, acc[nt], false, false);
        }
    }

    // epilogue: C/D layout VGPR r -> (M = r + half*8, N = l16)
#pragma unroll
    for (int nt = 0; nt < NT; ++nt) {
        const int n = nb + nt * 16 + l16;
        const float bn = bias ? bias[n] : 0.f;
#pragma unroll
        for (int r = 0; r < 8; ++r) {
            const int mm = blockIdx.x * 16 + r + (half << 3);
            float v = act_f(acc[nt][r] + bn, act);
            if (Res) v += Res[mm * ldr + n];
            Out[mm * ldo + n] = v;
        }
    }
}

// ---------------------------------------------------------------------------
// LayerNorm over C=256 (one 256-thread block per token row)
// ---------------------------------------------------------------------------
__global__ void k_ln(const float* __restrict__ in, const float* __restrict__ g,
                     const float* __restrict__ b, float* __restrict__ out, float eps)
{
    __shared__ float red[256];
    const int r = blockIdx.x, t = threadIdx.x;
    float x = in[r * C_DIM + t];
    red[t] = x; __syncthreads();
    for (int s = 128; s > 0; s >>= 1) { if (t < s) red[t] += red[t + s]; __syncthreads(); }
    float mean = red[0] * (1.f / C_DIM);
    __syncthreads();
    float d = x - mean;
    red[t] = d * d; __syncthreads();
    for (int s = 128; s > 0; s >>= 1) { if (t < s) red[t] += red[t + s]; __syncthreads(); }
    float var = red[0] * (1.f / C_DIM);
    out[r * C_DIM + t] = d * rsqrtf(var + eps) * g[t] + b[t];
}

// out = x[...,off:off+3] @ w^T + b (+ spos[n])
__global__ void k_embed(const float* __restrict__ x6, const float* __restrict__ w,
                        const float* __restrict__ b, const float* __restrict__ pos,
                        float* __restrict__ out, int off)
{
    const int r = blockIdx.x, c = threadIdx.x;
    const float* p = x6 + r * 6 + off;
    float v = w[c * 3] * p[0] + w[c * 3 + 1] * p[1] + w[c * 3 + 2] * p[2] + b[c];
    if (pos) v += pos[(r % NJ_DIM) * C_DIM + c];
    out[r * C_DIM + c] = v;
}

// view transpose [f,n,C] <-> [n,f,C], optional +tpos on s->t
__global__ void k_rearr(const float* __restrict__ in, float* __restrict__ out,
                        const float* __restrict__ pos, int s2t)
{
    const int r = blockIdx.x, c = threadIdx.x;
    if (s2t) {                               // out row r = n*F + f
        int n = r / F_DIM, f = r % F_DIM;
        float v = in[(f * NJ_DIM + n) * C_DIM + c];
        if (pos) v += pos[f * C_DIM + c];
        out[r * C_DIM + c] = v;
    } else {                                 // out row r = f*NJ + n
        int f = r / NJ_DIM, n = r % NJ_DIM;
        out[r * C_DIM + c] = in[(n * F_DIM + f) * C_DIM + c];
    }
}

// depthwise causal conv (K=4) + SiLU; flip=1 processes time-reversed sequence
__global__ void k_conv(const float* __restrict__ xin, int lda,
                       const float* __restrict__ w, const float* __restrict__ b,
                       float* __restrict__ out, int L, int flip)
{
    const int r = blockIdx.x, d = threadIdx.x;     // block = 512 threads
    const int seq = r / L, t = r % L;
    float acc = b[d];
#pragma unroll
    for (int k = 0; k < 4; ++k) {
        int j = t - 3 + k;
        if (j >= 0) {
            int row = seq * L + (flip ? (L - 1 - j) : j);
            acc += w[d * 4 + k] * xin[row * lda + d];
        }
    }
    out[r * DIN + d] = acc / (1.f + __expf(-acc));
}

// selective scan: one thread per (seq, channel); h[16] in regs; B/C in LDS
__global__ void k_scan(const float* __restrict__ xc, const float* __restrict__ delta,
                       const float* __restrict__ dbl, const float* __restrict__ A_log,
                       const float* __restrict__ Dp, float* __restrict__ y, int L)
{
    const int seq = blockIdx.x >> 1;
    const int d   = ((blockIdx.x & 1) << 8) + threadIdx.x;
    __shared__ float sBC[32];
    float h[DSTATE], An[DSTATE];
#pragma unroll
    for (int n = 0; n < DSTATE; ++n) { h[n] = 0.f; An[n] = -__expf(A_log[d * DSTATE + n]); }
    const float Dd = Dp[d];
    for (int t = 0; t < L; ++t) {
        const int r = seq * L + t;
        if (threadIdx.x < 32) sBC[threadIdx.x] = dbl[r * 48 + 16 + threadIdx.x]; // B|C
        __syncthreads();
        float dl = delta[r * DIN + d];
        float x  = xc[r * DIN + d];
        float dx = dl * x;
        float acc = 0.f;
#pragma unroll
        for (int n = 0; n < DSTATE; ++n) {
            h[n] = __expf(dl * An[n]) * h[n] + dx * sBC[n];
            acc += h[n] * sBC[16 + n];
        }
        y[r * DIN + d] = acc + x * Dd;
        __syncthreads();
    }
}

// y = 0.5*(yf + flip(yb)) * silu(z)
__global__ void k_combine(const float* __restrict__ yf, const float* __restrict__ yb,
                          const float* __restrict__ xz, float* __restrict__ yc, int L)
{
    const int r = blockIdx.x, d = threadIdx.x;     // block = 512
    const int seq = r / L, t = r % L;
    const int rb = seq * L + (L - 1 - t);
    float z = xz[r * 1024 + 512 + d];
    float s = z / (1.f + __expf(-z));
    yc[r * DIN + d] = 0.5f * (yf[r * DIN + d] + yb[rb * DIN + d]) * s;
}

// alpha = softmax([xm,lm] @ fw^T + fb); out = a0*xm + a1*lm
__global__ void k_fusion(const float* __restrict__ xm, const float* __restrict__ lm,
                         const float* __restrict__ fw, const float* __restrict__ fb,
                         float* __restrict__ out)
{
    __shared__ float r0[256], r1[256];
    __shared__ float a0s, a1s;
    const int r = blockIdx.x, t = threadIdx.x;
    float xv = xm[r * C_DIM + t], lv = lm[r * C_DIM + t];
    r0[t] = fw[t] * xv + fw[256 + t] * lv;
    r1[t] = fw[512 + t] * xv + fw[768 + t] * lv;
    __syncthreads();
    for (int s = 128; s > 0; s >>= 1) {
        if (t < s) { r0[t] += r0[t + s]; r1[t] += r1[t + s]; }
        __syncthreads();
    }
    if (t == 0) {
        float l0 = r0[0] + fb[0], l1 = r1[0] + fb[1];
        float mx = fmaxf(l0, l1);
        float e0 = __expf(l0 - mx), e1 = __expf(l1 - mx);
        a0s = e0 / (e0 + e1); a1s = e1 / (e0 + e1);
    }
    __syncthreads();
    out[r * C_DIM + t] = a0s * xv + a1s * lv;
}

// head: out[r,j] = xn[r] . head_w[j] + head_b[j]   (ln already applied to xn)
__global__ void k_head(const float* __restrict__ xn, const float* __restrict__ hw,
                       const float* __restrict__ hb, float* __restrict__ out, int T)
{
    int idx = blockIdx.x * blockDim.x + threadIdx.x;
    if (idx >= T * 3) return;
    int r = idx / 3, j = idx % 3;
    float acc = hb[j];
    for (int c = 0; c < C_DIM; ++c) acc += xn[r * C_DIM + c] * hw[j * C_DIM + c];
    out[idx] = acc;
}

// ---------------------------------------------------------------------------
// Host-side orchestration
// ---------------------------------------------------------------------------
struct MambaDirP { const float *conv_w, *conv_b, *xproj_w, *dt_w, *dt_b, *A_log, *D; };
struct BlockP {
    const float *n1g, *n1b, *n2g, *n2b;
    const float *in_w, *out_w;
    MambaDirP dir[2];
    const float *w1, *b1, *w2, *b2;
};
struct Bufs {
    float *X, *TMP1, *XZ, *XCF, *XCB, *DBLF, *DBLB, *DF, *DB, *YF, *YB, *YC, *H1, *X2;
    __bf16 *ABF, *WBF;
};

static BlockP get_block(const float* const* in, int base) {
    BlockP bp;
    bp.n1g = in[base + 0]; bp.n1b = in[base + 1]; bp.n2g = in[base + 2]; bp.n2b = in[base + 3];
    bp.in_w = in[base + 4]; bp.out_w = in[base + 5];
    for (int s = 0; s < 2; ++s) {
        int o = base + 6 + s * 7;
        bp.dir[s].conv_w  = in[o + 0]; bp.dir[s].conv_b = in[o + 1];
        bp.dir[s].xproj_w = in[o + 2]; bp.dir[s].dt_w   = in[o + 3];
        bp.dir[s].dt_b    = in[o + 4]; bp.dir[s].A_log  = in[o + 5];
        bp.dir[s].D       = in[o + 6];
    }
    bp.w1 = in[base + 20]; bp.b1 = in[base + 21];
    bp.w2 = in[base + 22]; bp.b2 = in[base + 23];
    return bp;
}

// Out[T,N] = act(A[:, :K] @ W[N,K]^T + bias) + Res.  K padded to 32 with zeros.
static void gemm(hipStream_t st, const Bufs& B,
                 const float* A, int lda, int K,
                 const float* W, int N,
                 const float* bias, const float* res, int ldr,
                 float* out, int ldo, int act)
{
    const int Kp = (K + 31) & ~31;
    k_cvt<<<T_PAD, 256, 0, st>>>(A, lda, B.ABF, Kp, K);   // activations -> bf16
    k_cvt<<<N,     256, 0, st>>>(W, K,   B.WBF, Kp, K);   // weights     -> bf16
    if ((N & 63) == 0) {
        dim3 g(M_TILES, (unsigned)(N / 64));
        k_gemm_bf<4><<<g, dim3(32), 0, st>>>(B.ABF, Kp, B.WBF, Kp, bias, res, ldr,
                                             out, ldo, Kp, act);
    } else {
        dim3 g(M_TILES, (unsigned)(N / 16));
        k_gemm_bf<1><<<g, dim3(32), 0, st>>>(B.ABF, Kp, B.WBF, Kp, bias, res, ldr,
                                             out, ldo, Kp, act);
    }
}

static void run_block(hipStream_t st, const BlockP& bp, const Bufs& B, int nseq, int L)
{
    const int T = nseq * L;   // 1377 in both views
    // t1 = ln(x)
    k_ln<<<T, C_DIM, 0, st>>>(B.X, bp.n1g, bp.n1b, B.TMP1, 1e-6f);
    // xz = t1 @ in_w^T   [T,1024]
    gemm(st, B, B.TMP1, C_DIM, C_DIM, bp.in_w, 1024, nullptr, nullptr, 0, B.XZ, 1024, 0);
    // causal convs (+silu) on xin = xz[:, :512]
    k_conv<<<T, DIN, 0, st>>>(B.XZ, 1024, bp.dir[0].conv_w, bp.dir[0].conv_b, B.XCF, L, 0);
    k_conv<<<T, DIN, 0, st>>>(B.XZ, 1024, bp.dir[1].conv_w, bp.dir[1].conv_b, B.XCB, L, 1);
    // dbl = xc @ xproj^T  [T,48]
    gemm(st, B, B.XCF, DIN, DIN, bp.dir[0].xproj_w, 48, nullptr, nullptr, 0, B.DBLF, 48, 0);
    gemm(st, B, B.XCB, DIN, DIN, bp.dir[1].xproj_w, 48, nullptr, nullptr, 0, B.DBLB, 48, 0);
    // delta = softplus(dbl[:, :16] @ dt_w^T + dt_b)  [T,512]  (K=16 zero-padded to 32)
    gemm(st, B, B.DBLF, 48, 16, bp.dir[0].dt_w, DIN, bp.dir[0].dt_b, nullptr, 0, B.DF, DIN, 3);
    gemm(st, B, B.DBLB, 48, 16, bp.dir[1].dt_w, DIN, bp.dir[1].dt_b, nullptr, 0, B.DB, DIN, 3);
    // selective scans
    k_scan<<<nseq * 2, 256, 0, st>>>(B.XCF, B.DF, B.DBLF, bp.dir[0].A_log, bp.dir[0].D, B.YF, L);
    k_scan<<<nseq * 2, 256, 0, st>>>(B.XCB, B.DB, B.DBLB, bp.dir[1].A_log, bp.dir[1].D, B.YB, L);
    // y = 0.5*(yf + flip(yb)) * silu(z)
    k_combine<<<T, DIN, 0, st>>>(B.YF, B.YB, B.XZ, B.YC, L);
    // x2 = y @ out_w^T + x
    gemm(st, B, B.YC, DIN, DIN, bp.out_w, C_DIM, nullptr, B.X, C_DIM, B.X2, C_DIM, 0);
    // MLP: x = gelu(ln(x2) @ w1^T + b1) @ w2^T + b2 + x2
    k_ln<<<T, C_DIM, 0, st>>>(B.X2, bp.n2g, bp.n2b, B.TMP1, 1e-6f);
    gemm(st, B, B.TMP1, C_DIM, C_DIM, bp.w1, DIN, bp.b1, nullptr, 0, B.H1, DIN, 2);
    gemm(st, B, B.H1, DIN, DIN, bp.w2, C_DIM, bp.b2, B.X2, C_DIM, B.X, C_DIM, 0);
}

extern "C" void kernel_launch(void* const* d_in, const int* in_sizes, int n_in,
                              void* d_out, int out_size, void* d_ws, size_t ws_size,
                              hipStream_t stream)
{
    (void)in_sizes; (void)n_in; (void)out_size; (void)ws_size;
    const float* const* in = (const float* const*)d_in;
    float* ws = (float*)d_ws;

    // input dict flattening (setup_inputs order)
    const int I_X = 0, I_E1W = 1, I_E1B = 2, I_E2W = 3, I_E2B = 4, I_SPOS = 5, I_TPOS = 6;
    const int I_STE0 = 7, I_STE1 = 31, I_TTE0 = 55, I_TTE1 = 79, I_LIMBS = 103, I_LIMBT = 127;
    const int I_SNG = 151, I_SNB = 152, I_TNG = 153, I_TNB = 154;
    const int I_LSG = 155, I_LSB = 156, I_LTG = 157, I_LTB = 158;
    const int I_FW = 159, I_FB = 160, I_HNG = 161, I_HNB = 162, I_HW = 163, I_HB = 164;

    // workspace layout (fp32 region, then bf16 region)
    size_t off = 0;
    Bufs B;
    B.X    = ws + off; off += (size_t)T_PAD * C_DIM;
    B.TMP1 = ws + off; off += (size_t)T_PAD * C_DIM;
    B.XZ   = ws + off; off += (size_t)T_PAD * 1024;
    B.XCF  = ws + off; off += (size_t)T_PAD * DIN;
    B.XCB  = ws + off; off += (size_t)T_PAD * DIN;
    B.DBLF = ws + off; off += (size_t)T_PAD * 48;
    B.DBLB = ws + off; off += (size_t)T_PAD * 48;
    B.DF   = ws + off; off += (size_t)T_PAD * DIN;
    B.DB   = ws + off; off += (size_t)T_PAD * DIN;
    B.YF   = ws + off; off += (size_t)T_PAD * DIN;
    B.YB   = ws + off; off += (size_t)T_PAD * DIN;
    B.YC   = ws + off; off += (size_t)T_PAD * DIN;
    B.H1   = ws + off; off += (size_t)T_PAD * DIN;
    B.X2   = ws + off; off += (size_t)T_PAD * C_DIM;
    float* XM = ws + off; off += (size_t)T_PAD * C_DIM;
    float* LM = ws + off; off += (size_t)T_PAD * C_DIM;
    B.ABF  = (__bf16*)(ws + off); off += (size_t)T_PAD * DIN / 2 + 64;   // T_PAD x 512 bf16
    B.WBF  = (__bf16*)(ws + off); off += (size_t)1024 * 256 / 2 + 64;    // 1024 x 256 bf16 max

    BlockP ste0 = get_block(in, I_STE0), ste1 = get_block(in, I_STE1);
    BlockP tte0 = get_block(in, I_TTE0), tte1 = get_block(in, I_TTE1);
    BlockP lbs  = get_block(in, I_LIMBS), lbt = get_block(in, I_LIMBT);

    const int T = T_TOK;

    // ---- joint (xyz) path ----
    k_embed<<<T, C_DIM, 0, stream>>>(in[I_X], in[I_E1W], in[I_E1B], in[I_SPOS], B.X, 0);
    run_block(stream, ste0, B, F_DIM, NJ_DIM);
    k_ln<<<T, C_DIM, 0, stream>>>(B.X, in[I_SNG], in[I_SNB], B.TMP1, 1e-6f);
    k_rearr<<<T, C_DIM, 0, stream>>>(B.TMP1, B.X, in[I_TPOS], 1);        // s->t (+tpos)
    run_block(stream, tte0, B, NJ_DIM, F_DIM);
    k_ln<<<T, C_DIM, 0, stream>>>(B.X, in[I_TNG], in[I_TNB], B.TMP1, 1e-6f);
    k_rearr<<<T, C_DIM, 0, stream>>>(B.TMP1, XM, nullptr, 0);            // t->s

    // ---- limb path ----
    k_embed<<<T, C_DIM, 0, stream>>>(in[I_X], in[I_E2W], in[I_E2B], nullptr, B.X, 3);
    run_block(stream, lbs, B, F_DIM, NJ_DIM);
    k_ln<<<T, C_DIM, 0, stream>>>(B.X, in[I_LSG], in[I_LSB], B.TMP1, 1e-6f);
    k_rearr<<<T, C_DIM, 0, stream>>>(B.TMP1, B.X, nullptr, 1);           // s->t (no tpos)
    run_block(stream, lbt, B, NJ_DIM, F_DIM);
    k_ln<<<T, C_DIM, 0, stream>>>(B.X, in[I_LTG], in[I_LTB], B.TMP1, 1e-6f);
    k_rearr<<<T, C_DIM, 0, stream>>>(B.TMP1, LM, nullptr, 0);            // t->s

    // ---- fusion ----
    k_fusion<<<T, C_DIM, 0, stream>>>(XM, LM, in[I_FW], in[I_FB], B.X);

    // ---- depth-1 spatio-temporal blocks ----
    run_block(stream, ste1, B, F_DIM, NJ_DIM);
    k_ln<<<T, C_DIM, 0, stream>>>(B.X, in[I_SNG], in[I_SNB], B.TMP1, 1e-6f);
    k_rearr<<<T, C_DIM, 0, stream>>>(B.TMP1, B.X, nullptr, 1);
    run_block(stream, tte1, B, NJ_DIM, F_DIM);
    k_ln<<<T, C_DIM, 0, stream>>>(B.X, in[I_TNG], in[I_TNB], B.TMP1, 1e-6f);
    k_rearr<<<T, C_DIM, 0, stream>>>(B.TMP1, B.X, nullptr, 0);

    // ---- head ----
    k_ln<<<T, C_DIM, 0, stream>>>(B.X, in[I_HNG], in[I_HNB], B.TMP1, 1e-5f);
    k_head<<<(T * 3 + 255) / 256, 256, 0, stream>>>(B.TMP1, in[I_HW], in[I_HB],
                                                    (float*)d_out, T);
}
